// trajectory_RNN_simple_14551349199554
// MI455X (gfx1250) — matrix operand order repaired and checked by reading, not compiled, not run on hardware
//
#include <hip/hip_runtime.h>
#include <hip/hip_bf16.h>
#include <math.h>

typedef __attribute__((ext_vector_type(16))) __bf16 v16bf;
typedef __attribute__((ext_vector_type(8)))  __bf16 v8bf;
typedef __attribute__((ext_vector_type(8)))  float  v8f;

#define NB    4096   // batch
#define NT    200    // time steps
#define NF    64     // sequence features
#define NA    128    // attribute features
#define NH    100    // hidden
#define HP    112    // padded per-gate width (7 tiles of 16)
#define KH    128    // padded hidden K dim
#define NTL   7      // hidden tiles per gate
#define NG    4      // gates i,f,g,o

// workspace byte offsets (all 16B aligned)
#define OFF_WKT 0u         // [4][7][16][64]  bf16  = 57344 B
#define OFF_WRT 57344u     // [4][7][16][128] bf16  = 114688 B
#define OFF_BLP 172032u    // [4][112] f32          = 1792 B
#define OFF_H   173824u    // [4096][112] f32       = 1835008 B
#define OFF_AT  2008832u   // [4096][100] f32       = 1638400 B
#define OFF_Y1  3647232u   // [4096][100] f32       = 1638400 B
#define OFF_Y2  5285632u   // [4096][50]  f32       = 819200 B

__device__ __forceinline__ v8f splat8(float x) {
  v8f v;
#pragma unroll
  for (int i = 0; i < 8; ++i) v[i] = x;
  return v;
}

// ---------------------------------------------------------------------------
// Pad + transpose LSTM weights into per-lane-contiguous bf16 B-fragment layout:
//   WkT[g][tile][n][kf]  (K contiguous, K=64)
//   WrT[g][tile][n][kh]  (K contiguous, K=128, rows >=100 zero)
//   blP[g][col]          (f32, cols >=100 zero)
// ---------------------------------------------------------------------------
__global__ void prep_weights(const float* __restrict__ Wk,
                             const float* __restrict__ Wr,
                             const float* __restrict__ bl,
                             __bf16* __restrict__ WkT,
                             __bf16* __restrict__ WrT,
                             float* __restrict__ blP) {
  int idx = blockIdx.x * blockDim.x + threadIdx.x;
  if (idx < NG * NTL * 16 * NF) {
    int kf = idx % NF, n = (idx / NF) & 15, tl = (idx / (NF * 16)) % NTL, g = idx / (NF * 16 * NTL);
    int col = tl * 16 + n;
    float v = (col < NH) ? Wk[kf * (4 * NH) + g * NH + col] : 0.f;
    WkT[idx] = (__bf16)v;
  }
  if (idx < NG * NTL * 16 * KH) {
    int kh = idx % KH, n = (idx / KH) & 15, tl = (idx / (KH * 16)) % NTL, g = idx / (KH * 16 * NTL);
    int col = tl * 16 + n;
    float v = (col < NH && kh < NH) ? Wr[kh * (4 * NH) + g * NH + col] : 0.f;
    WrT[idx] = (__bf16)v;
  }
  if (idx < NG * HP) {
    int col = idx % HP, g = idx / HP;
    blP[idx] = (col < NH) ? bl[g * NH + col] : 0.f;
  }
}

// ---------------------------------------------------------------------------
// Fused LSTM: 256 blocks x 16 batch rows, 7 waves (one hidden tile each).
// Weights live in VGPRs; c in fp32 registers; h double-buffered bf16 in LDS;
// x_t double-buffered fp32 in LDS via async global->LDS copies (wave 0).
// 24 v_wmma_f32_16x16x32_bf16 per wave per step.
// ---------------------------------------------------------------------------
__launch_bounds__(224, 1)
__global__ void lstm_fused(const float* __restrict__ seq,   // [B][T][F]
                           const __bf16* __restrict__ WkT,
                           const __bf16* __restrict__ WrT,
                           const float* __restrict__ blP,
                           float* __restrict__ hOut) {      // [B][HP]
  __shared__ float  xbuf[2][16][NF];   // 8 KB
  __shared__ __bf16 hbuf[2][16][KH];   // 8 KB
  const int tid = threadIdx.x;
  const int wv  = tid >> 5;     // 0..6  (hidden tile)
  const int ln  = tid & 31;
  const int r0  = blockIdx.x << 4;
  const int n   = ln & 15;      // B-frag column / A-frag row
  const int kh2 = ln >> 4;      // K half select

  // zero both h buffers (cols 112..127 must stay 0 forever)
  for (int i = tid; i < 2 * 16 * KH; i += 224) ((__bf16*)hbuf)[i] = (__bf16)0.f;

  // ---- preload all B fragments into registers (constant over T) ----
  // B order used here (same as A): lane holds col n, 16 contiguous K at kh2*16.
  v16bf bx[NG][2], bh[NG][4];
#pragma unroll
  for (int g = 0; g < NG; ++g) {
    const __bf16* p1 = WkT + ((g * NTL + wv) * 16 + n) * NF;
#pragma unroll
    for (int k = 0; k < 2; ++k) bx[g][k] = *(const v16bf*)(p1 + k * 32 + kh2 * 16);
    const __bf16* p2 = WrT + ((g * NTL + wv) * 16 + n) * KH;
#pragma unroll
    for (int k = 0; k < 4; ++k) bh[g][k] = *(const v16bf*)(p2 + k * 32 + kh2 * 16);
  }

  float bias[NG];
#pragma unroll
  for (int g = 0; g < NG; ++g) bias[g] = blP[g * HP + wv * 16 + n];

  v8f cst = splat8(0.f);   // cell state tile (C/D f32 layout)
  float hreg[8];
#pragma unroll
  for (int r = 0; r < 8; ++r) hreg[r] = 0.f;

  const unsigned xlds0 = (unsigned)(size_t)(&xbuf[0][0][0]);

  // prologue: async-copy x_0 into buffer 0
  if (wv == 0) {
#pragma unroll
    for (int j = 0; j < 8; ++j) {
      int chunk = j * 32 + ln;                 // 256 x 16B chunks
      int m = chunk >> 4, cw = chunk & 15;
      unsigned long long ga =
          (unsigned long long)(size_t)(seq + (size_t)(r0 + m) * NT * NF + (size_t)cw * 4);
      unsigned la = xlds0 + (unsigned)chunk * 16u;
      asm volatile("global_load_async_to_lds_b128 %0, %1, off" :: "v"(la), "v"(ga) : "memory");
    }
#if __has_builtin(__builtin_amdgcn_s_wait_asynccnt)
    __builtin_amdgcn_s_wait_asynccnt(0);
#else
    asm volatile("s_wait_asynccnt 0x0" ::: "memory");
#endif
  }
  __syncthreads();

  for (int t = 0; t < NT; ++t) {
    const int cur = t & 1, nx = cur ^ 1;

    // overlap: async-prefetch x_{t+1} into the other buffer
    if (wv == 0 && (t + 1) < NT) {
      unsigned xldsn = xlds0 + (unsigned)(nx * 16 * NF * 4);
#pragma unroll
      for (int j = 0; j < 8; ++j) {
        int chunk = j * 32 + ln;
        int m = chunk >> 4, cw = chunk & 15;
        unsigned long long ga = (unsigned long long)(size_t)(
            seq + (size_t)(r0 + m) * NT * NF + (size_t)(t + 1) * NF + (size_t)cw * 4);
        unsigned la = xldsn + (unsigned)chunk * 16u;
        asm volatile("global_load_async_to_lds_b128 %0, %1, off" :: "v"(la), "v"(ga) : "memory");
      }
    }

    v8f z[NG];
#pragma unroll
    for (int g = 0; g < NG; ++g) z[g] = splat8(bias[g]);

    // x part: K=64 (2 chunks). A order used here (same as B): lane = row n, halves
    // 0..15 -> K = 32k + kh2*16 .. +15 (contiguous).
#pragma unroll
    for (int k = 0; k < 2; ++k) {
      const float* xp = &xbuf[cur][n][k * 32 + kh2 * 16];
      v16bf a;
#pragma unroll
      for (int j = 0; j < 8; ++j) a[j] = (__bf16)xp[j];
#pragma unroll
      for (int j = 0; j < 8; ++j) a[8 + j] = (__bf16)xp[8 + j];
#pragma unroll
      for (int g = 0; g < NG; ++g)
        z[g] = __builtin_amdgcn_wmma_f32_16x16x32_bf16(false, a, false, bx[g][k],
                                                       (short)0, z[g], false, false);
    }
    // h part: K=128 (4 chunks) from LDS bf16
#pragma unroll
    for (int k = 0; k < 4; ++k) {
      const __bf16* hp = &hbuf[cur][n][k * 32 + kh2 * 16];
      v8bf lo = *(const v8bf*)hp;
      v8bf hi = *(const v8bf*)(hp + 8);
      v16bf a = __builtin_shufflevector(lo, hi, 0,1,2,3,4,5,6,7,8,9,10,11,12,13,14,15);
#pragma unroll
      for (int g = 0; g < NG; ++g)
        z[g] = __builtin_amdgcn_wmma_f32_16x16x32_bf16(false, a, false, bh[g][k],
                                                       (short)0, z[g], false, false);
    }

    // gate math on C/D f32 layout: vgpr r, lane -> (row kh2*8+r, col n)
#pragma unroll
    for (int r = 0; r < 8; ++r) {
      float i  = 1.f / (1.f + __expf(-z[0][r]));
      float f  = 1.f / (1.f + __expf(-z[1][r]));
      float g2 = fmaxf(z[2][r], 0.f);
      float o  = 1.f / (1.f + __expf(-z[3][r]));
      float c  = f * cst[r] + i * g2;
      cst[r]   = c;
      float h  = o * fmaxf(c, 0.f);
      hreg[r]  = h;
      hbuf[nx][kh2 * 8 + r][wv * 16 + n] = (__bf16)h;
    }

    if (wv == 0) {
#if __has_builtin(__builtin_amdgcn_s_wait_asynccnt)
      __builtin_amdgcn_s_wait_asynccnt(0);
#else
      asm volatile("s_wait_asynccnt 0x0" ::: "memory");
#endif
    }
    __syncthreads();
  }

  // final hidden state, fp32
#pragma unroll
  for (int r = 0; r < 8; ++r) {
    int m = kh2 * 8 + r;
    hOut[(size_t)(r0 + m) * HP + wv * 16 + n] = hreg[r];
  }
}

// ---------------------------------------------------------------------------
// Tiny head layers (sub-GFLOP): plain fp32 kernels, weights served from L2.
// ---------------------------------------------------------------------------
__global__ void dense_attr(const float* __restrict__ attr, const float* __restrict__ Wd,
                           const float* __restrict__ bd, float* __restrict__ out) {
  int idx = blockIdx.x * blockDim.x + threadIdx.x;
  if (idx >= NB * NH) return;
  int b = idx / NH, j = idx % NH;
  const float* a = attr + (size_t)b * NA;
  float s = bd[j];
  for (int k = 0; k < NA; ++k) s = fmaf(a[k], Wd[k * NH + j], s);
  out[idx] = fmaxf(s, 0.f);
}

__global__ void dense_p1(const float* __restrict__ at, const float* __restrict__ hA,
                         const float* __restrict__ W1, const float* __restrict__ b1,
                         float* __restrict__ out) {
  int idx = blockIdx.x * blockDim.x + threadIdx.x;
  if (idx >= NB * NH) return;
  int b = idx / NH, j = idx % NH;
  const float* ap = at + (size_t)b * NH;
  const float* hp = hA + (size_t)b * HP;
  float s = b1[j];
  for (int k = 0; k < NH; ++k) s = fmaf(ap[k], W1[k * NH + j], s);
  for (int k = 0; k < NH; ++k) s = fmaf(hp[k], W1[(NH + k) * NH + j], s);
  out[idx] = fmaxf(s, 0.f);
}

__global__ void dense_p2(const float* __restrict__ y1, const float* __restrict__ W2,
                         const float* __restrict__ b2, float* __restrict__ out) {
  int idx = blockIdx.x * blockDim.x + threadIdx.x;
  if (idx >= NB * 50) return;
  int b = idx / 50, j = idx % 50;
  const float* yp = y1 + (size_t)b * NH;
  float s = b2[j];
  for (int k = 0; k < NH; ++k) s = fmaf(yp[k], W2[k * 50 + j], s);
  out[idx] = fmaxf(s, 0.f);
}

__global__ void classify_softmax(const float* __restrict__ y2, const float* __restrict__ Wc,
                                 const float* __restrict__ bc, float* __restrict__ out) {
  int b = blockIdx.x * blockDim.x + threadIdx.x;
  if (b >= NB) return;
  const float* yp = y2 + (size_t)b * 50;
  float lg[10];
#pragma unroll
  for (int j = 0; j < 10; ++j) {
    float s = bc[j];
    for (int k = 0; k < 50; ++k) s = fmaf(yp[k], Wc[k * 10 + j], s);
    lg[j] = s;
  }
  float mx = lg[0];
#pragma unroll
  for (int j = 1; j < 10; ++j) mx = fmaxf(mx, lg[j]);
  float den = 0.f;
#pragma unroll
  for (int j = 0; j < 10; ++j) { lg[j] = __expf(lg[j] - mx); den += lg[j]; }
  float inv = 1.f / den;
#pragma unroll
  for (int j = 0; j < 10; ++j) out[(size_t)b * 10 + j] = lg[j] * inv;
}

extern "C" void kernel_launch(void* const* d_in, const int* in_sizes, int n_in,
                              void* d_out, int out_size, void* d_ws, size_t ws_size,
                              hipStream_t stream) {
  const float* attributes = (const float*)d_in[0];
  const float* sequences  = (const float*)d_in[1];
  const float* Wd = (const float*)d_in[2];
  const float* bd = (const float*)d_in[3];
  const float* Wk = (const float*)d_in[4];
  const float* Wr = (const float*)d_in[5];
  const float* bl = (const float*)d_in[6];
  const float* W1 = (const float*)d_in[7];
  const float* b1 = (const float*)d_in[8];
  const float* W2 = (const float*)d_in[9];
  const float* b2 = (const float*)d_in[10];
  const float* Wc = (const float*)d_in[11];
  const float* bc = (const float*)d_in[12];
  float* out = (float*)d_out;

  char* ws = (char*)d_ws;
  __bf16* WkT = (__bf16*)(ws + OFF_WKT);
  __bf16* WrT = (__bf16*)(ws + OFF_WRT);
  float*  blP = (float*)(ws + OFF_BLP);
  float*  hO  = (float*)(ws + OFF_H);
  float*  atB = (float*)(ws + OFF_AT);
  float*  y1  = (float*)(ws + OFF_Y1);
  float*  y2  = (float*)(ws + OFF_Y2);

  prep_weights<<<(NG * NTL * 16 * KH + 255) / 256, 256, 0, stream>>>(Wk, Wr, bl, WkT, WrT, blP);
  lstm_fused<<<NB / 16, 224, 0, stream>>>(sequences, WkT, WrT, blP, hO);
  dense_attr<<<(NB * NH + 255) / 256, 256, 0, stream>>>(attributes, Wd, bd, atB);
  dense_p1<<<(NB * NH + 255) / 256, 256, 0, stream>>>(atB, hO, W1, b1, y1);
  dense_p2<<<(NB * 50 + 255) / 256, 256, 0, stream>>>(y1, W2, b2, y2);
  classify_softmax<<<(NB + 255) / 256, 256, 0, stream>>>(y2, Wc, bc, out);
}
